// TransformerBlock_8074538516546
// MI455X (gfx1250) — matrix operand order repaired and checked by reading, not compiled, not run on hardware
//
#include <hip/hip_runtime.h>
#include <hip/hip_bf16.h>
#include <math.h>

typedef __attribute__((ext_vector_type(16))) __bf16 v16bf;
typedef __attribute__((ext_vector_type(8)))  __bf16 v8bf;
typedef __attribute__((ext_vector_type(4)))  __bf16 v4bf;
typedef __attribute__((ext_vector_type(8)))  float  v8f;

#define DIM 1024
#define SEQ 1024
#define BATCH 2
#define MTOT (BATCH*SEQ)      // 2048 rows
#define HEADS 16
#define HD 64
#define MLP_DIM 4096

__device__ __forceinline__ v8f wmma_bf16(v16bf a, v16bf b, v8f c) {
    // D = A(16x32 bf16) * B(32x16 bf16) + C(16x16 f32)
    return __builtin_amdgcn_wmma_f32_16x16x32_bf16(
        false, a, false, b, (short)0, c, false, false);
}

// Fragment layout (ISA 7.12.2, 16-bit A 16x32 / B 32x16):
// lane L: row/col = L&15, kbase = (L>>4)*8; elems 0-7 -> K=kbase..+7,
// elems 8-15 -> K=kbase+16..+23.  Both halves are 16B-contiguous in LDS.
__device__ __forceinline__ v16bf load_frag_lds(const __bf16* p0, const __bf16* p1) {
    v8bf lo = *(const v8bf*)p0;
    v8bf hi = *(const v8bf*)p1;
    v16bf r;
#pragma unroll
    for (int j = 0; j < 8; ++j) { r[j] = lo[j]; r[j + 8] = hi[j]; }
    return r;
}

// ---------------- LayerNorm: one block per row of 1024 ----------------
__global__ __launch_bounds__(256) void ln_kernel(
    const float* __restrict__ x, const float* __restrict__ g,
    const float* __restrict__ b, float* __restrict__ y)
{
    __shared__ float red[256];
    const int row = blockIdx.x;
    const int tid = threadIdx.x;
    const float* xr = x + (size_t)row * DIM;

    float s = 0.f;
#pragma unroll
    for (int i = 0; i < 4; ++i) s += xr[tid + i * 256];
    red[tid] = s; __syncthreads();
#pragma unroll
    for (int o = 128; o > 0; o >>= 1) { if (tid < o) red[tid] += red[tid + o]; __syncthreads(); }
    const float mu = red[0] * (1.f / DIM);
    __syncthreads();

    float v = 0.f;
#pragma unroll
    for (int i = 0; i < 4; ++i) { float d = xr[tid + i * 256] - mu; v += d * d; }
    red[tid] = v; __syncthreads();
#pragma unroll
    for (int o = 128; o > 0; o >>= 1) { if (tid < o) red[tid] += red[tid + o]; __syncthreads(); }
    const float rstd = rsqrtf(red[0] * (1.f / DIM) + 1e-5f);

    float* yr = y + (size_t)row * DIM;
#pragma unroll
    for (int i = 0; i < 4; ++i) {
        int c = tid + i * 256;
        yr[c] = (xr[c] - mu) * rstd * g[c] + b[c];
    }
}

// ---------------- Generic bf16-WMMA GEMM, fp32 in/out ----------------
// C[M,N] = epilogue( A[M,K] @ B[K,N] ).  128x128 tile per 256-thread block
// (8 waves, each owns a 32x64 sub-block = 8 WMMA tiles), BK=32.
// Double-buffered LDS with register-staged prefetch: next tile's global
// loads issue before the current tile's WMMAs, so XDL work hides latency.
#define BM 128
#define BN 128
#define BK 32
template<bool HAS_BIAS, bool HAS_RES, bool DO_GELU>
__global__ __launch_bounds__(256) void gemm_wmma(
    const float* __restrict__ A, const float* __restrict__ B,
    const float* __restrict__ bias, const float* __restrict__ residual,
    float* __restrict__ C, int M, int N, int K)
{
    __shared__ __bf16 sA[2][BM][BK];   // 2 x 8 KB, row-major [m][k]
    __shared__ __bf16 sB[2][BN][BK];   // 2 x 8 KB, transposed [n][k]

    const int tid  = threadIdx.x;
    const int wave = tid >> 5;
    const int lane = tid & 31;
    const int mblk = blockIdx.y * BM;
    const int nblk = blockIdx.x * BN;

    const int mrow  = lane & 15;
    const int kgrp  = lane >> 4;
    const int kbase = kgrp * 8;
    const int wrow  = (wave >> 1) * 32;   // 0,32,64,96
    const int wcol  = (wave & 1) * 64;    // 0,64

    // per-thread staging slots (fixed): A -> (r, c-quad), B -> (n, k-quad)
    const int aR = tid >> 3,        aC = (tid & 7) * 4;        // + it*32 rows
    const int bN = tid & 127,       bK = (tid >> 7) * 4;       // + it*8 k

    float4 aReg[4];
    float  bReg[4][4];

    auto loadTile = [&](int k0) {
#pragma unroll
        for (int it = 0; it < 4; ++it)      // rows aR + it*32
            aReg[it] = *(const float4*)&A[(size_t)(mblk + aR + it * 32) * K + k0 + aC];
#pragma unroll
        for (int it = 0; it < 4; ++it) {    // k-quads bK + it*8
            int k = bK + it * 8;
#pragma unroll
            for (int j = 0; j < 4; ++j)
                bReg[it][j] = B[(size_t)(k0 + k + j) * N + nblk + bN];
        }
    };
    auto storeTile = [&](int buf) {
#pragma unroll
        for (int it = 0; it < 4; ++it) {
            v4bf pk = { (__bf16)aReg[it].x, (__bf16)aReg[it].y,
                        (__bf16)aReg[it].z, (__bf16)aReg[it].w };
            *(v4bf*)&sA[buf][aR + it * 32][aC] = pk;
        }
#pragma unroll
        for (int it = 0; it < 4; ++it) {
            v4bf pk = { (__bf16)bReg[it][0], (__bf16)bReg[it][1],
                        (__bf16)bReg[it][2], (__bf16)bReg[it][3] };
            *(v4bf*)&sB[buf][bN][bK + it * 8] = pk;
        }
    };

    v8f acc[2][4] = {};

    loadTile(0);
    storeTile(0);
    __syncthreads();

    int buf = 0;
    for (int k0 = 0; k0 < K; k0 += BK) {
        const bool more = (k0 + BK) < K;
        if (more) loadTile(k0 + BK);        // global loads in flight during WMMA

        v16bf afrag[2];
#pragma unroll
        for (int mt = 0; mt < 2; ++mt)
            afrag[mt] = load_frag_lds(&sA[buf][wrow + mt * 16 + mrow][kbase],
                                      &sA[buf][wrow + mt * 16 + mrow][kbase + 16]);
        v16bf bfrag[4];
#pragma unroll
        for (int nt = 0; nt < 4; ++nt) {
            int ncol = wcol + nt * 16 + mrow;
            bfrag[nt] = load_frag_lds(&sB[buf][ncol][kbase], &sB[buf][ncol][kbase + 16]);
        }
#pragma unroll
        for (int mt = 0; mt < 2; ++mt)
#pragma unroll
            for (int nt = 0; nt < 4; ++nt)
                acc[mt][nt] = wmma_bf16(afrag[mt], bfrag[nt], acc[mt][nt]);

        if (more) storeTile(buf ^ 1);       // waits on loads only here
        __syncthreads();
        buf ^= 1;
    }

    // epilogue + store (C elem r: row = r + kgrp*8, col = lane&15)
#pragma unroll
    for (int mt = 0; mt < 2; ++mt) {
        int rowbase = mblk + wrow + mt * 16 + kgrp * 8;
#pragma unroll
        for (int nt = 0; nt < 4; ++nt) {
            int col = nblk + wcol + nt * 16 + mrow;
            float bv = HAS_BIAS ? bias[col] : 0.f;
#pragma unroll
            for (int r = 0; r < 8; ++r) {
                int row = rowbase + r;
                float v = acc[mt][nt][r] + bv;
                if (DO_GELU) v = 0.5f * v * (1.f + erff(v * 0.70710678118f));
                if (HAS_RES) v += residual[(size_t)row * N + col];
                C[(size_t)row * N + col] = v;
            }
        }
    }
}

// ---------------- Attention: one wave per (batch, head, 16-row tile) ----------------
// qkv layout: [batch*seq, 3072] with col = which*1024 + head*64 + d
__global__ __launch_bounds__(32) void attn_kernel(
    const float* __restrict__ qkv, float* __restrict__ out)
{
    __shared__ __bf16 sc[16][SEQ];  // 32 KB unnormalized-prob buffer
    __shared__ __bf16 sV[HD][32];   // 4 KB V chunk, transposed [n][k]
    __shared__ float  sInv[16];     // per-row 1/sum

    const int rt   = blockIdx.x;       // query row tile 0..63
    const int h    = blockIdx.y;       // head
    const int bb   = blockIdx.z;       // batch
    const int lane = threadIdx.x;
    const int mrow = lane & 15;
    const int kgrp = lane >> 4;
    const int kbase = kgrp * 8;

    const float* base = qkv + (size_t)bb * SEQ * (3 * DIM);
    const int qoff = 0 * DIM + h * HD;
    const int koff = 1 * DIM + h * HD;
    const int voff = 2 * DIM + h * HD;
    const float scale = 0.03125f;      // DIM^-0.5 = 1/32

    // Q fragments: head-dim 64 -> two K=32 chunks (contiguous 8-float runs)
    v16bf qa[2];
    {
        const float* qrow = base + (size_t)(rt * 16 + mrow) * (3 * DIM) + qoff;
#pragma unroll
        for (int c = 0; c < 2; ++c) {
            int kb = c * 32 + kbase;
#pragma unroll
            for (int j = 0; j < 8; ++j) {
                qa[c][j]     = (__bf16)qrow[kb + j];
                qa[c][j + 8] = (__bf16)qrow[kb + 16 + j];
            }
        }
    }

    // scores = Q @ K^T * scale  -> LDS (bf16)
    for (int kt = 0; kt < SEQ / 16; ++kt) {
        const float* krow = base + (size_t)(kt * 16 + mrow) * (3 * DIM) + koff;
        v8f s = {};
#pragma unroll
        for (int c = 0; c < 2; ++c) {
            int kb = c * 32 + kbase;
            v16bf bfrag;
#pragma unroll
            for (int j = 0; j < 8; ++j) {
                bfrag[j]     = (__bf16)krow[kb + j];
                bfrag[j + 8] = (__bf16)krow[kb + 16 + j];
            }
            s = wmma_bf16(qa[c], bfrag, s);
        }
#pragma unroll
        for (int r = 0; r < 8; ++r)
            sc[r + kgrp * 8][kt * 16 + mrow] = (__bf16)(s[r] * scale);
    }
    __syncthreads();

    // softmax (2 lanes per row, 512 cols each): vectorized v8bf scans,
    // branch-free max, exp stored unnormalized; 1/sum folded into epilogue.
    {
        const int row  = lane >> 1;
        __bf16* rp = &sc[row][(lane & 1) * 512];

        float mx = -3.4e38f;
        for (int i = 0; i < 512; i += 8) {
            v8bf v = *(const v8bf*)&rp[i];
#pragma unroll
            for (int j = 0; j < 8; ++j) {
                float f = (float)v[j];
                mx = f > mx ? f : mx;
            }
        }
        float mo = __shfl_xor(mx, 1);
        mx = mo > mx ? mo : mx;

        float sum = 0.f;
        for (int i = 0; i < 512; i += 8) {
            v8bf v = *(const v8bf*)&rp[i];
            v8bf e;
#pragma unroll
            for (int j = 0; j < 8; ++j) {
                float f = __expf((float)v[j] - mx);
                e[j] = (__bf16)f;
                sum += f;
            }
            *(v8bf*)&rp[i] = e;
        }
        sum += __shfl_xor(sum, 1);
        if ((lane & 1) == 0) sInv[row] = 1.f / sum;
    }
    __syncthreads();

    float invr[8];
#pragma unroll
    for (int r = 0; r < 8; ++r) invr[r] = sInv[kbase + r];

    // out = P(16x1024) @ V(1024x64); V staged per 32-token chunk,
    // coalesced global reads (lane = feature col), transposed into LDS.
    v8f o[4] = {};
    for (int kc = 0; kc < SEQ / 32; ++kc) {
        int kb = kc * 32;
#pragma unroll
        for (int c = 0; c < 2; ++c) {
            int n = lane + 32 * c;                  // feature column 0..63
            const float* vcol = base + voff + n;
#pragma unroll
            for (int r = 0; r < 32; r += 4) {
                v4bf pk;
#pragma unroll
                for (int j = 0; j < 4; ++j)
                    pk[j] = (__bf16)vcol[(size_t)(kb + r + j) * (3 * DIM)];
                *(v4bf*)&sV[n][r] = pk;
            }
        }
        __syncthreads();

        v16bf pa = load_frag_lds(&sc[mrow][kb + kbase], &sc[mrow][kb + kbase + 16]);
#pragma unroll
        for (int nt = 0; nt < 4; ++nt) {
            int ncol = nt * 16 + mrow;
            v16bf vb = load_frag_lds(&sV[ncol][kbase], &sV[ncol][kbase + 16]);
            o[nt] = wmma_bf16(pa, vb, o[nt]);
        }
        __syncthreads();
    }

    // store (softmax normalization applied here): out[b, token, h*64 + col]
    float* ob = out + (size_t)bb * SEQ * DIM;
#pragma unroll
    for (int nt = 0; nt < 4; ++nt) {
#pragma unroll
        for (int r = 0; r < 8; ++r) {
            int tok = rt * 16 + r + kgrp * 8;
            int col = h * HD + nt * 16 + mrow;
            ob[(size_t)tok * DIM + col] = o[nt][r] * invr[r];
        }
    }
}

// ---------------- host driver ----------------
extern "C" void kernel_launch(void* const* d_in, const int* in_sizes, int n_in,
                              void* d_out, int out_size, void* d_ws, size_t ws_size,
                              hipStream_t stream) {
    const float* x     = (const float*)d_in[0];
    const float* ln1_g = (const float*)d_in[1];
    const float* ln1_b = (const float*)d_in[2];
    const float* w_qkv = (const float*)d_in[3];
    const float* w_o   = (const float*)d_in[4];
    const float* b_o   = (const float*)d_in[5];
    const float* ln2_g = (const float*)d_in[6];
    const float* ln2_b = (const float*)d_in[7];
    const float* w1    = (const float*)d_in[8];
    const float* b1    = (const float*)d_in[9];
    const float* w2    = (const float*)d_in[10];
    const float* b2    = (const float*)d_in[11];

    float* xbuf = (float*)d_out;                         // running residual stream
    float* ws   = (float*)d_ws;
    float* xn    = ws;                                   // 2M f32
    float* qkvb  = ws + (size_t)MTOT * DIM;              // 6M f32
    float* attnb = qkvb + (size_t)MTOT * (3 * DIM);      // 2M f32
    float* hdn   = qkvb;                                 // alias (qkv/attn dead in MLP phase)

    hipMemcpyAsync(xbuf, x, (size_t)MTOT * DIM * sizeof(float),
                   hipMemcpyDeviceToDevice, stream);

    for (int layer = 0; layer < 4; ++layer) {
        // attention block
        ln_kernel<<<MTOT, 256, 0, stream>>>(xbuf, ln1_g, ln1_b, xn);
        gemm_wmma<false, false, false><<<dim3((3 * DIM) / BN, MTOT / BM), 256, 0, stream>>>(
            xn, w_qkv, nullptr, nullptr, qkvb, MTOT, 3 * DIM, DIM);
        attn_kernel<<<dim3(SEQ / 16, HEADS, BATCH), 32, 0, stream>>>(qkvb, attnb);
        gemm_wmma<true, true, false><<<dim3(DIM / BN, MTOT / BM), 256, 0, stream>>>(
            attnb, w_o, b_o, xbuf, xbuf, MTOT, DIM, DIM);
        // MLP block
        ln_kernel<<<MTOT, 256, 0, stream>>>(xbuf, ln2_g, ln2_b, xn);
        gemm_wmma<true, false, true><<<dim3(MLP_DIM / BN, MTOT / BM), 256, 0, stream>>>(
            xn, w1, b1, nullptr, hdn, MTOT, MLP_DIM, DIM);
        gemm_wmma<true, true, false><<<dim3(DIM / BN, MTOT / BM), 256, 0, stream>>>(
            hdn, w2, b2, xbuf, xbuf, MTOT, DIM, MLP_DIM);
    }
}
